// MultiHeadAttention_67156108640886
// MI455X (gfx1250) — matrix operand order repaired and checked by reading, not compiled, not run on hardware
//
#include <hip/hip_runtime.h>
#include <hip/hip_bf16.h>
#include <stdint.h>

// Problem constants (from reference): B=2, S=2048, D=1024, H=16, DH=64
#define B_SZ    2
#define S_LEN   2048
#define D_MODEL 1024
#define H_NUM   16
#define D_HEAD  64

typedef __bf16 bf16t;
typedef __attribute__((ext_vector_type(8)))  __bf16 v8bf;
typedef __attribute__((ext_vector_type(16))) __bf16 v16bf;
typedef __attribute__((ext_vector_type(8)))  float  v8f;
typedef __attribute__((ext_vector_type(4)))  int    i32x4;

#if __has_builtin(__builtin_amdgcn_global_load_async_to_lds_b128)
#define HAS_ASYNC_LDS 1
#else
#define HAS_ASYNC_LDS 0
#endif

// 16-byte global -> LDS copy; async (ASYNCcnt) when the builtin exists.
__device__ __forceinline__ void async_copy16(const void* g, void* l) {
#if HAS_ASYNC_LDS
  __builtin_amdgcn_global_load_async_to_lds_b128(
      (__attribute__((address_space(1))) i32x4*)g,
      (__attribute__((address_space(3))) i32x4*)l, 0, 0);
#else
  *(int4*)l = *(const int4*)g;
#endif
}

__device__ __forceinline__ void async_wait0() {
#if HAS_ASYNC_LDS
#if __has_builtin(__builtin_amdgcn_s_wait_asynccnt)
  __builtin_amdgcn_s_wait_asynccnt(0);
#else
  asm volatile("s_wait_asynccnt 0x0" ::: "memory");
#endif
#endif
}

// Concatenate two 16-byte (8 x bf16) runs into one WMMA 16-element operand.
__device__ __forceinline__ v16bf concat8(const bf16t* lo, const bf16t* hi) {
  v8bf a = *(const v8bf*)lo;
  v8bf b = *(const v8bf*)hi;
  v16bf r;
#pragma unroll
  for (int i = 0; i < 8; ++i) { r[i] = a[i]; r[i + 8] = b[i]; }
  return r;
}

__device__ __forceinline__ v8f wmma_bf16(v16bf a, v16bf b, v8f c) {
  return __builtin_amdgcn_wmma_f32_16x16x32_bf16(false, a, false, b, (short)0, c,
                                                 false, false);
}

__device__ __forceinline__ v8f zero8f() {
  v8f z;
#pragma unroll
  for (int i = 0; i < 8; ++i) z[i] = 0.0f;
  return z;
}

// ---------------------------------------------------------------------------
// One-time weight prep: W fp32 [K,N] -> WT bf16 [N,K] (convert + transpose).
// Grid (N/64, K/64), 128 threads.
// ---------------------------------------------------------------------------
__global__ void __launch_bounds__(128)
wcvt_transpose(const float* __restrict__ W, bf16t* __restrict__ WT)
{
  constexpr int K = D_MODEL, N = D_MODEL;
  __shared__ bf16t T[64][72];          // T[k][n]
  const int tid = threadIdx.x;
  const int n0 = blockIdx.x * 64, k0 = blockIdx.y * 64;
#pragma unroll
  for (int i = tid; i < 64 * 16; i += 128) {     // 64 k-rows x 16 float4
    int r = i >> 4, c = (i & 15) << 2;
    float4 t = *(const float4*)&W[(size_t)(k0 + r) * N + n0 + c];
    T[r][c + 0] = (bf16t)t.x; T[r][c + 1] = (bf16t)t.y;
    T[r][c + 2] = (bf16t)t.z; T[r][c + 3] = (bf16t)t.w;
  }
  __syncthreads();
#pragma unroll
  for (int i = tid; i < 64 * 8; i += 128) {      // 64 n-rows x 8 v8bf
    int n = i >> 3, c = (i & 7) << 3;
    v8bf o;
#pragma unroll
    for (int j = 0; j < 8; ++j) o[j] = T[c + j][n];
    *(v8bf*)&WT[(size_t)(n0 + n) * K + k0 + c] = o;
  }
}

// ---------------------------------------------------------------------------
// 64x64-tile bf16 WMMA GEMM:  C = A @ W + bias   (M=B*S, K=N=1024)
// WT is bf16 [N,K] (pre-transposed) -> B tiles staged by pure async copies.
//   IN_HEADS=0: A fp32 [M,K] (VALU-converted)  IN_HEADS=1: A bf16 [B,H,S,DH]
//   OUT_F32=0 : C -> bf16 [B,H,S,DH]           OUT_F32=1 : C -> fp32 [M,N]
// ---------------------------------------------------------------------------
template <int IN_HEADS, int OUT_F32>
__global__ void __launch_bounds__(128)
gemm_wmma(const void* __restrict__ Aptr, const bf16t* __restrict__ WT,
          const float* __restrict__ bias, void* __restrict__ Cptr)
{
  constexpr int K = D_MODEL, N = D_MODEL;
  __shared__ __align__(16) bf16t As[64][40];   // [m][k], row stride 80B
  __shared__ __align__(16) bf16t Bs[64][40];   // [n][k]

  const int tid  = threadIdx.x;
  const int wave = tid >> 5;
  const int lane = tid & 31;
  const int lm   = lane & 15;
  const int kh   = lane >> 4;
  const int row0 = blockIdx.y * 64;
  const int col0 = blockIdx.x * 64;

  v8f acc[4];
#pragma unroll
  for (int nt = 0; nt < 4; ++nt) acc[nt] = zero8f();

  for (int kk = 0; kk < K; kk += 32) {
    __syncthreads();
    if (IN_HEADS == 0) {
      const float* A = (const float*)Aptr;
#pragma unroll
      for (int i = tid; i < 64 * 8; i += 128) {        // 64 rows x 8 float4
        int r = i >> 3, c = (i & 7) << 2;
        float4 t = *(const float4*)&A[(size_t)(row0 + r) * K + kk + c];
        As[r][c + 0] = (bf16t)t.x; As[r][c + 1] = (bf16t)t.y;
        As[r][c + 2] = (bf16t)t.z; As[r][c + 3] = (bf16t)t.w;
      }
    } else {
      const bf16t* A = (const bf16t*)Aptr;             // [B,H,S,DH] bf16
      const int h = kk >> 6;
#pragma unroll
      for (int i = tid; i < 256; i += 128) {           // 64 rows x 4 x 16B
        int r = i >> 2, c = (i & 3) << 3;
        int m = row0 + r, b = m >> 11, s = m & (S_LEN - 1);
        async_copy16(&A[(((size_t)(b * H_NUM + h)) * S_LEN + s) * D_HEAD
                        + (kk & 63) + c],
                     &As[r][c]);
      }
    }
    // B tile: contiguous rows of WT -> async 16B chunks, no VALU
#pragma unroll
    for (int i = tid; i < 256; i += 128) {             // 64 n-rows x 4 x 16B
      int r = i >> 2, c = (i & 3) << 3;
      async_copy16(&WT[(size_t)(col0 + r) * K + kk + c], &Bs[r][c]);
    }
    async_wait0();
    __syncthreads();

    const int arow = wave * 16 + lm;
    v16bf a = concat8(&As[arow][8 * kh], &As[arow][16 + 8 * kh]);
#pragma unroll
    for (int nt = 0; nt < 4; ++nt) {
      v16bf b = concat8(&Bs[nt * 16 + lm][8 * kh],
                        &Bs[nt * 16 + lm][16 + 8 * kh]);
      acc[nt] = wmma_bf16(a, b, acc[nt]);
    }
  }

#pragma unroll
  for (int nt = 0; nt < 4; ++nt) {
#pragma unroll
    for (int r = 0; r < 8; ++r) {
      int m = row0 + wave * 16 + r + 8 * kh;           // C-layout row
      int n = col0 + nt * 16 + lm;                     // C-layout col
      float cv = acc[nt][r] + bias[n];
      if (OUT_F32) {
        ((float*)Cptr)[(size_t)m * (size_t)N + n] = cv;
      } else {
        int b = m >> 11, s = m & (S_LEN - 1), h = n >> 6, dh = n & 63;
        ((bf16t*)Cptr)[(((size_t)(b * H_NUM + h)) * S_LEN + s) * D_HEAD + dh] =
            (bf16t)cv;
      }
    }
  }
}

// ---------------------------------------------------------------------------
// Flash attention: one block = 64 query rows of one (b,h); 4 waves x 16 rows.
// Mask tile staged via async global->LDS b128; V staged transposed in LDS;
// Q/K fragments loaded straight from global; P reshaped via per-wave LDS.
// ---------------------------------------------------------------------------
__global__ void __launch_bounds__(128)
attn_wmma(const bf16t* __restrict__ Q, const bf16t* __restrict__ Km,
          const bf16t* __restrict__ V, const uint8_t* __restrict__ mask,
          bf16t* __restrict__ O)
{
  __shared__ __align__(16) bf16t   Vt[64][80];     // Vt[d][key], stride 160B
  __shared__ __align__(16) bf16t   Ps[4][16][80];  // per-wave P [m][key]
  __shared__ __align__(16) uint8_t Ms[64][64];     // mask tile [qrow][key]

  const int tid  = threadIdx.x;
  const int wave = tid >> 5;
  const int lane = tid & 31;
  const int lm   = lane & 15;
  const int kh   = lane >> 4;

  constexpr int NT = S_LEN / 64;                   // 32 tiles
  const int qt = blockIdx.x % NT;
  const int bh = blockIdx.x / NT;                  // 0..B*H-1
  const size_t hoff = (size_t)bh * S_LEN * D_HEAD;
  const bf16t* Qh = Q  + hoff;
  const bf16t* Kh = Km + hoff;
  const bf16t* Vh = V  + hoff;
  const uint8_t* Mh = mask + (size_t)bh * S_LEN * S_LEN;

  // Persistent Q A-fragments for DH chunks 0..31 / 32..63
  const int qrow = qt * 64 + wave * 16 + lm;
  v16bf qf[2];
#pragma unroll
  for (int c = 0; c < 2; ++c)
    qf[c] = concat8(&Qh[(size_t)qrow * D_HEAD + c * 32 + 8 * kh],
                    &Qh[(size_t)qrow * D_HEAD + c * 32 + 16 + 8 * kh]);

  v8f oacc[4];
#pragma unroll
  for (int dt = 0; dt < 4; ++dt) oacc[dt] = zero8f();
  float rmax[8], rsum[8];
#pragma unroll
  for (int r = 0; r < 8; ++r) { rmax[r] = -1.0e30f; rsum[r] = 0.0f; }

  const float scale = 0.125f;                      // 1/sqrt(64)

  for (int kt = 0; kt < NT; ++kt) {
    const int kbase = kt * 64;
    __syncthreads();
    // Mask tile: 64 rows x 64B, async 16B chunks
#pragma unroll
    for (int i = tid; i < 256; i += 128) {
      int r = i >> 2, c = (i & 3) << 4;
      async_copy16(&Mh[(size_t)(qt * 64 + r) * S_LEN + kbase + c], &Ms[r][c]);
    }
    // V tile transposed: Vt[d][n]
#pragma unroll
    for (int i = tid; i < 64 * 8; i += 128) {
      int n = i >> 3, d0 = (i & 7) << 3;
      v8bf t = *(const v8bf*)&Vh[(size_t)(kbase + n) * D_HEAD + d0];
#pragma unroll
      for (int j = 0; j < 8; ++j) Vt[d0 + j][n] = t[j];
    }
    async_wait0();
    __syncthreads();

    // Scores S = Q @ K^T : K rows ARE the B-fragments (contiguous along DH)
    v8f s[4];
#pragma unroll
    for (int nt = 0; nt < 4; ++nt) {
      s[nt] = zero8f();
      const int krow = kbase + nt * 16 + lm;
#pragma unroll
      for (int c = 0; c < 2; ++c) {
        v16bf kf = concat8(&Kh[(size_t)krow * D_HEAD + c * 32 + 8 * kh],
                           &Kh[(size_t)krow * D_HEAD + c * 32 + 16 + 8 * kh]);
        s[nt] = wmma_bf16(qf[c], kf, s[nt]);
      }
    }

    // Scale + mask from LDS (mask==true -> -inf)
#pragma unroll
    for (int nt = 0; nt < 4; ++nt) {
#pragma unroll
      for (int r = 0; r < 8; ++r) {
        float sv = s[nt][r] * scale;
        if (Ms[wave * 16 + r + 8 * kh][nt * 16 + lm] != 0) sv = -1.0e30f;
        s[nt][r] = sv;
      }
    }

    // Online softmax (row r lives in VGPR r across the lane's 16-lane half)
    float corr[8];
#pragma unroll
    for (int r = 0; r < 8; ++r) {
      float mx = fmaxf(fmaxf(s[0][r], s[1][r]), fmaxf(s[2][r], s[3][r]));
#pragma unroll
      for (int off = 1; off < 16; off <<= 1)
        mx = fmaxf(mx, __shfl_xor(mx, off, 32));
      float mnew = fmaxf(rmax[r], mx);
      corr[r] = __expf(rmax[r] - mnew);
      rmax[r] = mnew;
      float ps = 0.0f;
#pragma unroll
      for (int nt = 0; nt < 4; ++nt) {
        float p = __expf(s[nt][r] - mnew);
        s[nt][r] = p;
        ps += p;
      }
#pragma unroll
      for (int off = 1; off < 16; off <<= 1)
        ps += __shfl_xor(ps, off, 32);
      rsum[r] = rsum[r] * corr[r] + ps;
    }
#pragma unroll
    for (int dt = 0; dt < 4; ++dt)
#pragma unroll
      for (int r = 0; r < 8; ++r) oacc[dt][r] *= corr[r];

    // P: C-layout -> A-layout through per-wave LDS (in-order per wave)
#pragma unroll
    for (int nt = 0; nt < 4; ++nt)
#pragma unroll
      for (int r = 0; r < 8; ++r)
        Ps[wave][r + 8 * kh][nt * 16 + lm] = (bf16t)s[nt][r];

    // O += P @ V
#pragma unroll
    for (int c = 0; c < 2; ++c) {
      v16bf pf = concat8(&Ps[wave][lm][c * 32 + 8 * kh],
                         &Ps[wave][lm][c * 32 + 16 + 8 * kh]);
#pragma unroll
      for (int dt = 0; dt < 4; ++dt) {
        v16bf vf = concat8(&Vt[dt * 16 + lm][c * 32 + 8 * kh],
                           &Vt[dt * 16 + lm][c * 32 + 16 + 8 * kh]);
        oacc[dt] = wmma_bf16(pf, vf, oacc[dt]);
      }
    }
  }

  // Normalize and store bf16 [B,H,S,DH]
#pragma unroll
  for (int r = 0; r < 8; ++r) {
    float inv = rsum[r] > 0.0f ? 1.0f / rsum[r] : 0.0f;
    int m = qt * 64 + wave * 16 + r + 8 * kh;
#pragma unroll
    for (int dt = 0; dt < 4; ++dt)
      O[hoff + (size_t)m * D_HEAD + dt * 16 + lm] = (bf16t)(oacc[dt][r] * inv);
  }
}

// ---------------------------------------------------------------------------
extern "C" void kernel_launch(void* const* d_in, const int* in_sizes, int n_in,
                              void* d_out, int out_size, void* d_ws, size_t ws_size,
                              hipStream_t stream)
{
  const float*   q    = (const float*)d_in[0];
  const float*   k    = (const float*)d_in[1];
  const float*   v    = (const float*)d_in[2];
  const uint8_t* mask = (const uint8_t*)d_in[3];   // jnp bool = 1 byte/elem
  const float*   Wq   = (const float*)d_in[4];
  const float*   bq   = (const float*)d_in[5];
  const float*   Wk   = (const float*)d_in[6];
  const float*   bk   = (const float*)d_in[7];
  const float*   Wv   = (const float*)d_in[8];
  const float*   bv   = (const float*)d_in[9];
  const float*   Wo   = (const float*)d_in[10];
  const float*   bo   = (const float*)d_in[11];
  float* out = (float*)d_out;

  // Workspace layout (bf16):
  //   4 transposed weights [N,K]  : 4 x 1M elems  =  8 MB
  //   qh, kh, vh, ao [B,H,S,DH]   : 4 x 4M elems  = 32 MB
  const size_t wElems    = (size_t)D_MODEL * D_MODEL;          // 1M
  const size_t headElems = (size_t)B_SZ * H_NUM * S_LEN * D_HEAD;  // 4M
  bf16t* WTq = (bf16t*)d_ws;
  bf16t* WTk = WTq + wElems;
  bf16t* WTv = WTk + wElems;
  bf16t* WTo = WTv + wElems;
  bf16t* qh  = WTo + wElems;
  bf16t* kh  = qh + headElems;
  bf16t* vh  = kh + headElems;
  bf16t* ao  = vh + headElems;

  dim3 gt(D_MODEL / 64, D_MODEL / 64);             // (16, 16)
  wcvt_transpose<<<gt, 128, 0, stream>>>(Wq, WTq);
  wcvt_transpose<<<gt, 128, 0, stream>>>(Wk, WTk);
  wcvt_transpose<<<gt, 128, 0, stream>>>(Wv, WTv);
  wcvt_transpose<<<gt, 128, 0, stream>>>(Wo, WTo);

  dim3 gg(D_MODEL / 64, (B_SZ * S_LEN) / 64);      // (16, 64)
  gemm_wmma<0, 0><<<gg, 128, 0, stream>>>(q, WTq, bq, qh);
  gemm_wmma<0, 0><<<gg, 128, 0, stream>>>(k, WTk, bk, kh);
  gemm_wmma<0, 0><<<gg, 128, 0, stream>>>(v, WTv, bv, vh);

  attn_wmma<<<dim3(B_SZ * H_NUM * (S_LEN / 64)), 128, 0, stream>>>(
      qh, kh, vh, mask, ao);

  gemm_wmma<1, 1><<<gg, 128, 0, stream>>>(ao, WTo, bo, out);
}
